// LieConvLayer_3977139716372
// MI455X (gfx1250) — matrix operand order repaired and checked by reading, not compiled, not run on hardware
//
#include <hip/hip_runtime.h>
#include <hip/hip_bf16.h>
#include <math.h>

typedef unsigned short u16;
typedef unsigned int   u32;
typedef unsigned long long u64;

typedef __attribute__((ext_vector_type(16))) __bf16 v16bf;
typedef __attribute__((ext_vector_type(8)))  float  v8f;
typedef __attribute__((ext_vector_type(4)))  unsigned int v4u;
typedef __attribute__((ext_vector_type(8)))  int    v8i;
typedef __attribute__((ext_vector_type(4)))  int    v4i;

#define N_NODES 10000
#define N_EDGES 100000
#define D_      248
#define DP      256     // D padded to multiple of 32 for WMMA K-steps
#define H_      512
#define NNZ_    2480
#define LN_EPS  1e-5f

// ---------- helpers ----------

__device__ __forceinline__ u16 f2bf(float x) {
    union { float f; u32 u; } t; t.f = x;
    u32 u = t.u;
    if ((u & 0x7fffffffu) > 0x7f800000u) return (u16)((u >> 16) | 0x4000u); // NaN
    u32 r = u + 0x7fffu + ((u >> 16) & 1u);                                 // RNE
    return (u16)(r >> 16);
}

// A-matrix 16x32 bf16 tile per ISA layout (lane m=lane&15, half=lane>>4):
//  vec elems [2v,2v+1]: v<4 -> K = kb + half*8 + 2v ; v>=4 -> K = kb + 16 + half*8 + 2(v-4)
__device__ __forceinline__ v16bf loadA_bf16(const u16* base, int rowStride, int lane, int kb) {
    const int m = lane & 15, half = lane >> 4;
    const u16* row = base + m * rowStride;
    union { v16bf v; u32 u[8]; } a;
#pragma unroll
    for (int v = 0; v < 4; ++v) a.u[v]     = *(const u32*)(row + kb + half * 8 + 2 * v);
#pragma unroll
    for (int v = 0; v < 4; ++v) a.u[v + 4] = *(const u32*)(row + kb + 16 + half * 8 + 2 * v);
    return a.v;
}

// B-matrix 32x16 bf16 tile: B[k][n] = W[n][k] (W row-major over k).
// lane col n = ncol0 + (lane&15); same K order as A: v<4 -> K = kb + khalf*8 + 2v ; v>=4 -> K = kb + 16 + khalf*8 + 2(v-4)
__device__ __forceinline__ v16bf loadB_bf16(const u16* W, int rowStride, int ncol0, int lane, int kb) {
    const int n = ncol0 + (lane & 15), khalf = lane >> 4;
    const u16* row = W + n * rowStride;
    union { v16bf v; u32 u[8]; } b;
#pragma unroll
    for (int v = 0; v < 4; ++v) b.u[v]     = *(const u32*)(row + kb + khalf * 8 + 2 * v);
#pragma unroll
    for (int v = 0; v < 4; ++v) b.u[v + 4] = *(const u32*)(row + kb + 16 + khalf * 8 + 2 * v);
    return b.v;
}

__device__ __forceinline__ v8f wmma_bf16(v16bf a, v16bf b, v8f c) {
    return __builtin_amdgcn_wmma_f32_16x16x32_bf16(false, a, false, b, (short)0, c, false, false);
}

__device__ __forceinline__ float wred32(float v) {
#pragma unroll
    for (int o = 16; o > 0; o >>= 1) v += __shfl_xor(v, o, 32);
    return v;
}

// Tensor Data Mover: 1-D global -> LDS copy of `ndwords` dwords.
// D# per CDNA5 ISA ch.8: group0 {count=1 | lds_addr | global_addr(57b) | type=2},
// group1 {data_size=4B, tensor_dim0=tile_dim0=ndwords, tensor_dim1=tile_dim1=1,
//         tensor_dim0_stride=ndwords}; groups 2/3 zero (<=2D tensor).
// amdgpu-toolchain (clang-23) 6-arg form: (v4u, v8i, v4i, v4i, v8i, i32 cpol).
__device__ __forceinline__ void tdm_load_1d(u32 lds_addr, const void* gptr, u32 ndwords) {
    u64 ga = (u64)(uintptr_t)gptr;
    v4u g0;
    g0.x = 0x1u;                                           // count=1, user mode
    g0.y = lds_addr;                                       // LDS dest (bytes)
    g0.z = (u32)(ga & 0xffffffffu);                        // global_addr[31:0]
    g0.w = (u32)((ga >> 32) & 0x1ffffffu) | (2u << 30);    // global_addr[56:32] | type=2
    v8i g1;
    g1[0] = (int)(2u << 16);                               // data_size code 2 = 4 bytes
    g1[1] = (int)((ndwords & 0xffffu) << 16);              // tensor_dim0[15:0] -> bits 63:48
    g1[2] = (int)(((ndwords >> 16) & 0xffffu) | (1u << 16)); // tensor_dim0[31:16] | tensor_dim1=1
    g1[3] = (int)((ndwords & 0xffffu) << 16);              // tile_dim0 -> bits 127:112
    g1[4] = 1;                                             // tile_dim1 = 1
    g1[5] = (int)ndwords;                                  // tensor_dim0_stride[31:0]
    g1[6] = 0;
    g1[7] = 0;
    v4i z4 = {0, 0, 0, 0};
    v8i z8 = {0, 0, 0, 0, 0, 0, 0, 0};
    __builtin_amdgcn_tensor_load_to_lds(g0, g1, z4, z4, z8, 0);
}

// ---------- conversion kernels (f32 -> bf16, zero padded) ----------

__global__ __launch_bounds__(256) void k_cvt_pad(const float* __restrict__ src, u16* __restrict__ dst,
                                                 int R, int C, int Rp, int Cp) {
    int total = Rp * Cp;
    for (int i = blockIdx.x * 256 + threadIdx.x; i < total; i += gridDim.x * 256) {
        int r = i / Cp, c = i - r * Cp;
        float v = (r < R && c < C) ? src[r * C + c] : 0.0f;
        dst[i] = f2bf(v);
    }
}

// pack structure constants into one contiguous dword tensor for a single TDM copy:
// trip[t] = I|J<<8|K<<16 ; trip[NNZ+t] = bits(C[t])
__global__ __launch_bounds__(256) void k_pack_trip(const int* __restrict__ Iidx, const int* __restrict__ Jidx,
                                                   const int* __restrict__ Kidx, const float* __restrict__ Cval,
                                                   u32* __restrict__ trip) {
    int t = blockIdx.x * 256 + threadIdx.x;
    if (t < NNZ_) {
        trip[t] = (u32)Iidx[t] | ((u32)Jidx[t] << 8) | ((u32)Kidx[t] << 16);
        union { float f; u32 u; } c; c.f = Cval[t];
        trip[NNZ_ + t] = c.u;
    }
}

// ---------- fused edge kernel: gather + WMMA projection + Lie bracket + scatter ----------

__global__ __launch_bounds__(256) void k_edge(const float* __restrict__ feats,
                                              const int*   __restrict__ eidx,
                                              const u16*   __restrict__ featsBf,
                                              const u16*   __restrict__ wmsgBf,
                                              const u32*   __restrict__ tripG,
                                              float* __restrict__ agg) {
    __shared__ int sSrcIdx[16];
    __shared__ int sTgtIdx[16];
    __shared__ __align__(16) u16   sSrc[16 * DP];   // gathered src rows, bf16
    __shared__ __align__(16) float sTgt[16 * DP];   // gathered tgt rows, f32
    __shared__ __align__(16) float sProj[16 * DP];  // src_proj tile, f32
    __shared__ __align__(16) float sMsg[16 * DP];   // bracket output, f32
    __shared__ __align__(16) u32   sTrip[2 * NNZ_]; // packed triples + C bits (TDM dest)

    const int tid = threadIdx.x;
    const int e0  = blockIdx.x * 16;

    // wave 0 kicks off the TDM copy of the structure-constant table while
    // the rest of the workgroup stages the gathered rows.
    if (tid == 0) {
        tdm_load_1d((u32)(uintptr_t)&sTrip[0], tripG, 2 * NNZ_);
    }

    if (tid < 16) {
        sSrcIdx[tid] = eidx[e0 + tid];
        sTgtIdx[tid] = eidx[N_EDGES + e0 + tid];
    }
    __syncthreads();

    // stage gathered src rows (bf16, 16B vectors)
    for (int i = tid; i < 16 * DP / 8; i += 256) {
        int m = i >> 5, c8 = (i & 31) * 8;
        *(uint4*)&sSrc[m * DP + c8] = *(const uint4*)&featsBf[(size_t)sSrcIdx[m] * DP + c8];
    }
    // stage gathered tgt rows (f32), zero the pad + message tile
    for (int i = tid; i < 16 * DP; i += 256) {
        int m = i >> 8, k = i & 255;
        sTgt[i] = (k < D_) ? feats[(size_t)sTgtIdx[m] * D_ + k] : 0.0f;
        sMsg[i] = 0.0f;
    }
    if (tid == 0) {
        __builtin_amdgcn_s_wait_tensorcnt(0);   // triples landed in LDS
    }
    asm volatile("" ::: "memory");
    __syncthreads();

    // WMMA: src_proj(16xD) = A(16xK) * Wmsg^T ; 8 waves x 2 N-tiles
    const int w = tid >> 5, lane = tid & 31;
    const int nl = lane & 15, half = lane >> 4;
    const int nt0 = w * 2, nt1 = w * 2 + 1;
    v8f acc0 = {}, acc1 = {};
#pragma unroll
    for (int ks = 0; ks < 8; ++ks) {
        const int kb = ks * 32;
        v16bf a  = loadA_bf16(sSrc, DP, lane, kb);
        v16bf b0 = loadB_bf16(wmsgBf, DP, nt0 * 16, lane, kb);
        acc0 = wmma_bf16(a, b0, acc0);
        v16bf b1 = loadB_bf16(wmsgBf, DP, nt1 * 16, lane, kb);
        acc1 = wmma_bf16(a, b1, acc1);
    }
    {
        union { v8f v; float f[8]; } u0, u1; u0.v = acc0; u1.v = acc1;
#pragma unroll
        for (int r = 0; r < 8; ++r) {
            int m = r + half * 8;
            sProj[m * DP + nt0 * 16 + nl] = u0.f[r];
            sProj[m * DP + nt1 * 16 + nl] = u1.f[r];
        }
    }
    __syncthreads();

    // sparse Lie bracket: 16 threads per edge, LDS f32 atomics
    {
        const int e = tid >> 4, sub = tid & 15;
        const float* proj = &sProj[e * DP];
        const float* tg   = &sTgt[e * DP];
        float*       mg   = &sMsg[e * DP];
        for (int t = sub; t < NNZ_; t += 16) {
            u32 p = sTrip[t];
            union { u32 u; float f; } c; c.u = sTrip[NNZ_ + t];
            float v = c.f * proj[p & 255] * tg[(p >> 8) & 255];
            __hip_atomic_fetch_add(&mg[(p >> 16) & 255], v,
                                   __ATOMIC_RELAXED, __HIP_MEMORY_SCOPE_WORKGROUP);
        }
    }
    __syncthreads();

    // scatter-add message tile onto agg[tgt] (agent-scope f32 atomics)
    for (int i = tid; i < 16 * D_; i += 256) {
        int e = i / D_, d = i - e * D_;
        __hip_atomic_fetch_add(&agg[(size_t)sTgtIdx[e] * DP + d], sMsg[e * DP + d],
                               __ATOMIC_RELAXED, __HIP_MEMORY_SCOPE_AGENT);
    }
}

// ---------- fused MLP + residual + LayerNorm ----------

__global__ __launch_bounds__(256) void k_mlp_ln(const float* __restrict__ feats,
                                                const float* __restrict__ agg,
                                                const u16* __restrict__ w1Bf, const float* __restrict__ b1,
                                                const u16* __restrict__ w2Bf, const float* __restrict__ b2,
                                                const float* __restrict__ gamma, const float* __restrict__ beta,
                                                float* __restrict__ out) {
    __shared__ __align__(16) u16   sA[16 * DP];     // agg tile, bf16
    __shared__ __align__(16) u16   sHid[16 * H_];   // silu(hidden) tile, bf16
    __shared__ __align__(16) float sH[16 * DP];     // final MLP output, f32

    const int tid = threadIdx.x;
    const int n0  = blockIdx.x * 16;
    const int w = tid >> 5, lane = tid & 31;
    const int nl = lane & 15, half = lane >> 4;

    for (int i = tid; i < 16 * DP; i += 256) {
        int m = i >> 8, k = i & 255;
        sA[i] = f2bf(agg[(size_t)(n0 + m) * DP + k]);   // pad cols are already 0
    }
    __syncthreads();

    // GEMM1: hidden(16x512) = A(16x248) * W1^T ; 8 waves x 4 N-tiles; +bias, SiLU, ->bf16
    {
        v8f acc[4] = {{}, {}, {}, {}};
#pragma unroll
        for (int ks = 0; ks < 8; ++ks) {
            const int kb = ks * 32;
            v16bf a = loadA_bf16(sA, DP, lane, kb);
#pragma unroll
            for (int jj = 0; jj < 4; ++jj) {
                const int nt = w + 8 * jj;
                __builtin_prefetch(w1Bf + (size_t)(nt * 16 + nl) * DP + kb + 32, 0, 1);
                v16bf b = loadB_bf16(w1Bf, DP, nt * 16, lane, kb);
                acc[jj] = wmma_bf16(a, b, acc[jj]);
            }
        }
#pragma unroll
        for (int jj = 0; jj < 4; ++jj) {
            const int nt = w + 8 * jj;
            const int n  = nt * 16 + nl;
            const float bias = b1[n];
            union { v8f v; float f[8]; } u; u.v = acc[jj];
#pragma unroll
            for (int r = 0; r < 8; ++r) {
                int m = r + half * 8;
                float x = u.f[r] + bias;
                float s = x / (1.0f + __expf(-x));       // SiLU
                sHid[m * H_ + n] = f2bf(s);
            }
        }
    }
    __syncthreads();

    // GEMM2: h(16x248) = hidden(16x512) * W2^T ; 8 waves x 2 N-tiles; +bias -> f32 LDS
    {
        v8f d0 = {}, d1 = {};
        const int ntA = w, ntB = w + 8;
#pragma unroll
        for (int ks = 0; ks < 16; ++ks) {
            const int kb = ks * 32;
            v16bf a = loadA_bf16(sHid, H_, lane, kb);
            v16bf bA = loadB_bf16(w2Bf, H_, ntA * 16, lane, kb);
            d0 = wmma_bf16(a, bA, d0);
            v16bf bB = loadB_bf16(w2Bf, H_, ntB * 16, lane, kb);
            d1 = wmma_bf16(a, bB, d1);
        }
        union { v8f v; float f[8]; } uA, uB; uA.v = d0; uB.v = d1;
        {
            const int n = ntA * 16 + nl;      // n in [0,128): always valid
            const float bias = b2[n];
#pragma unroll
            for (int r = 0; r < 8; ++r) sH[(r + half * 8) * DP + n] = uA.f[r] + bias;
        }
        {
            const int n = ntB * 16 + nl;      // n in [128,256): guard pad
            if (n < D_) {
                const float bias = b2[n];
#pragma unroll
                for (int r = 0; r < 8; ++r) sH[(r + half * 8) * DP + n] = uB.f[r] + bias;
            }
        }
    }
    __syncthreads();

    // residual + LayerNorm: wave w handles rows 2w, 2w+1
#pragma unroll
    for (int rr = 0; rr < 2; ++rr) {
        const int m = w * 2 + rr;
        const int node = n0 + m;
        float xr[8];
        float sum = 0.0f;
        int cnt = 0;
        for (int d = lane; d < D_; d += 32, ++cnt) {
            float x = feats[(size_t)node * D_ + d] + sH[m * DP + d];
            xr[cnt] = x;
            sum += x;
        }
        sum = wred32(sum);
        const float mu = sum * (1.0f / (float)D_);
        float s2 = 0.0f;
        for (int c = 0; c < cnt; ++c) { float xc = xr[c] - mu; s2 += xc * xc; }
        s2 = wred32(s2);
        const float inv = rsqrtf(s2 * (1.0f / (float)D_) + LN_EPS);
        int c = 0;
        for (int d = lane; d < D_; d += 32, ++c) {
            out[(size_t)node * D_ + d] = (xr[c] - mu) * inv * gamma[d] + beta[d];
        }
    }
}

// ---------- host launcher ----------

extern "C" void kernel_launch(void* const* d_in, const int* in_sizes, int n_in,
                              void* d_out, int out_size, void* d_ws, size_t ws_size,
                              hipStream_t stream) {
    const float* feats = (const float*)d_in[0];
    const int*   eidx  = (const int*)  d_in[1];
    const float* Wmsg  = (const float*)d_in[2];
    const float* W1    = (const float*)d_in[3];
    const float* b1    = (const float*)d_in[4];
    const float* W2    = (const float*)d_in[5];
    const float* b2    = (const float*)d_in[6];
    const float* gamma = (const float*)d_in[7];
    const float* beta  = (const float*)d_in[8];
    const int*   Iidx  = (const int*)  d_in[9];
    const int*   Jidx  = (const int*)  d_in[10];
    const int*   Kidx  = (const int*)  d_in[11];
    const float* Cval  = (const float*)d_in[12];
    float*       out   = (float*)d_out;

    char* ws = (char*)d_ws;
    u16*   featsBf = (u16*)(ws);                                   // 10000*256*2 = 5,120,000
    u16*   wmsgBf  = (u16*)(ws + 5120000);                         // 256*256*2   =   131,072
    u16*   w1Bf    = (u16*)(ws + 5120000 + 131072);                // 512*256*2   =   262,144
    u16*   w2Bf    = (u16*)(ws + 5120000 + 131072 + 262144);       // 256*512*2   =   262,144
    float* agg     = (float*)(ws + 5120000 + 131072 + 262144 + 262144);     // 10000*256*4
    u32*   tripG   = (u32*)(ws + 5120000 + 131072 + 262144 + 262144 + 10240000); // 4960*4

    // precision conversions (bf16, zero-padded K) + triple packing
    k_cvt_pad<<<(N_NODES * DP + 255) / 256, 256, 0, stream>>>(feats, featsBf, N_NODES, D_, N_NODES, DP);
    k_cvt_pad<<<(DP * DP + 255) / 256, 256, 0, stream>>>(Wmsg, wmsgBf, D_, D_, DP, DP);
    k_cvt_pad<<<(H_ * DP + 255) / 256, 256, 0, stream>>>(W1, w1Bf, H_, D_, H_, DP);
    k_cvt_pad<<<(DP * H_ + 255) / 256, 256, 0, stream>>>(W2, w2Bf, D_, H_, DP, H_);
    k_pack_trip<<<(NNZ_ + 255) / 256, 256, 0, stream>>>(Iidx, Jidx, Kidx, Cval, tripG);

    (void)hipMemsetAsync(agg, 0, (size_t)N_NODES * DP * sizeof(float), stream);

    // fused edge pipeline: 6250 tiles of 16 edges
    k_edge<<<N_EDGES / 16, 256, 0, stream>>>(feats, eidx, featsBf, wmsgBf, tripG, agg);

    // fused MLP + residual + LayerNorm: 625 tiles of 16 nodes
    k_mlp_ln<<<N_NODES / 16, 256, 0, stream>>>(feats, agg, w1Bf, b1, w2Bf, b2,
                                               gamma, beta, out);
}
